// GINInteraction_66597762892471
// MI455X (gfx1250) — matrix operand order repaired and checked by reading, not compiled, not run on hardware
//
#include <hip/hip_runtime.h>

// ---------------------------------------------------------------------------
// CDNA5 (gfx1250, wave32) GIN interaction layer.
// All GEMMs use v_wmma_f32_16x16x32_bf16 (bf16 operands, fp32 accumulate).
// Round 4: compile-time bias-add guard (no +0.0 in MODE 0 epilogue); otherwise
// identical to round 3 (fragment-packed LDS weights, batched gathers,
// PHASE-templated edge kernel, 32-bit row offsets).
// ---------------------------------------------------------------------------

typedef __attribute__((ext_vector_type(16))) __bf16 v16bf;
typedef __attribute__((ext_vector_type(8)))  __bf16 v8bf;
typedef __attribute__((ext_vector_type(8)))  float  v8f;

#define D  128   // node feature dim (in == out == hidden)
#define DE 32    // edge feature dim

// ---- WMMA helpers ----------------------------------------------------------

__device__ __forceinline__ v8f wmma_bf16(v16bf a, v16bf b, v8f c) {
  // 8 args: (neg_a, A, neg_b, B, c_mod, C, reuse_a, reuse_b)
  return __builtin_amdgcn_wmma_f32_16x16x32_bf16(false, a, false, b,
                                                 (short)0, c, false, false);
}

// A-matrix 16x32 bf16 layout (ISA 7.12.2): lane l holds row m=l&15,
// khi=l>>4. Fragment element j maps to local K:
//   j=0..7  -> k0 + j        with k0 = 8*khi   (VGPR0-3 group)
//   j=8..15 -> k0 + 16 + j-8                   (VGPR4-7 group)
// i.e. two contiguous 8-element runs -> vectorizable loads.

__device__ __forceinline__ float fast_tanh(float x) {
#if __has_builtin(__builtin_amdgcn_tanhf)
  return __builtin_amdgcn_tanhf(x);   // v_tanh_f32 (TRANS op, co-executes)
#else
  float e = __expf(2.0f * x);
  return 1.0f - 2.0f / (e + 1.0f);
#endif
}

// Order-preserving mapping float -> uint32 so atomicMax works for any sign.
__device__ __forceinline__ unsigned enc_f32(float f) {
  unsigned u = __float_as_uint(f);
  return (u & 0x80000000u) ? ~u : (u | 0x80000000u);
}
__device__ __forceinline__ float dec_f32(unsigned k) {
  unsigned b = (k & 0x80000000u) ? (k & 0x7fffffffu) : ~k;
  return __uint_as_float(b);
}

// Load the 16 A-fragment source floats (two contiguous float4 pairs).
__device__ __forceinline__ void load16f(const float* __restrict__ p, int k0,
                                        float* __restrict__ f) {
  const float4 a = *(const float4*)(p + k0);
  const float4 b = *(const float4*)(p + k0 + 4);
  const float4 c = *(const float4*)(p + k0 + 16);
  const float4 d = *(const float4*)(p + k0 + 20);
  f[0] = a.x; f[1] = a.y; f[2]  = a.z; f[3]  = a.w;
  f[4] = b.x; f[5] = b.y; f[6]  = b.z; f[7]  = b.w;
  f[8] = c.x; f[9] = c.y; f[10] = c.z; f[11] = c.w;
  f[12] = d.x; f[13] = d.y; f[14] = d.z; f[15] = d.w;
}

// Pre-pack weight matrix W[K][D] (row-major, K = 32*KT) into per-fragment
// contiguous v16bf so the per-wave B load is two ds_load_b128.
// Fragment index f = (ct*KT + kt)*32 + lane; element j <-> K = kt*32+khi*8+(j<8?j:j+8),
// N = ct*16 + (lane&15).   (B 32x16 layout mirrored from ISA sparse-B table.)
template <int KT>
__device__ __forceinline__ void pack_weights(v16bf* __restrict__ wp,
                                             const float* __restrict__ W) {
  const int nfrag = 8 * KT * 32;
  for (int f = threadIdx.x; f < nfrag; f += blockDim.x) {
    const int lane = f & 31;
    const int kt   = (f >> 5) % KT;
    const int ct   = (f >> 5) / KT;
    const int khi  = lane >> 4, ln = lane & 15;
    const int n     = ct * 16 + ln;
    const int kbase = kt * 32 + khi * 8;
    v16bf v;
#pragma unroll
    for (int j = 0; j < 16; ++j) v[j] = (__bf16)W[(kbase + (j < 8 ? j : j + 8)) * D + n];
    wp[f] = v;
  }
}

// ---- init ------------------------------------------------------------------

__global__ void init_kernel(float* __restrict__ agg, float* __restrict__ denom,
                            unsigned* __restrict__ maxenc,
                            float* __restrict__ colsum, float* __restrict__ colsumsq,
                            int N) {
  int i = blockIdx.x * blockDim.x + threadIdx.x;
  if (i < N * D) agg[i] = 0.0f;
  if (i < N) { denom[i] = 0.0f; maxenc[i] = enc_f32(-3.402823466e38f); }
  if (i < D) { colsum[i] = 0.0f; colsumsq[i] = 0.0f; }
}

// ---- generic 128-K node GEMM: one 16-row M-tile / block, wave = col tile ---
// MODE 0: h   = A32 @ W                          (store fp32)
// MODE 1: t16 = bf16(tanh(((1+eps)A32+Aadd)@W+b))
// MODE 2: t16 = bf16(A16@W + b)
// MODE 3: z   = A16@W + b   (store fp32 + batch-norm column sum/sumsq atomics)
// MODE 4: out = tanh(A32*scale+shift)@W + b      (store fp32)

template <int MODE>
__global__ __launch_bounds__(256)
void gemm128_kernel(const float* __restrict__ A32, const __bf16* __restrict__ A16,
                    const float* __restrict__ Aadd, const float* __restrict__ W,
                    const float* __restrict__ bias,
                    const float* __restrict__ scale, const float* __restrict__ shift,
                    const float* __restrict__ epsp,
                    float* __restrict__ out32, __bf16* __restrict__ out16,
                    float* __restrict__ colsum, float* __restrict__ colsumsq) {
  __shared__ v16bf wlds[8 * 4 * 32];        // 32 KB, fragment-packed
  pack_weights<4>(wlds, W);
  __syncthreads();

  const int lane = threadIdx.x & 31;
  const int wave = threadIdx.x >> 5;        // 8 waves -> 8 col tiles of 16
  const int khi  = lane >> 4, ln = lane & 15;
  const unsigned row  = blockIdx.x * 16 + ln;   // A row owned by this lane
  const unsigned col  = wave * 16 + ln;         // output column of this lane
  const unsigned arow = row * D;                // 32-bit offsets (buffers < 4GB)

  float c1 = 1.0f;
  if (MODE == 1) c1 = 1.0f + epsp[0];

  v8f acc = {0.f, 0.f, 0.f, 0.f, 0.f, 0.f, 0.f, 0.f};
#pragma unroll
  for (int kt = 0; kt < 4; ++kt) {
    const int k0 = kt * 32 + 8 * khi;       // fragment's first local K
    v16bf a;
    if (MODE == 0 || MODE == 1 || MODE == 4) {
      float fa[16];
      load16f(A32 + arow, k0, fa);
      if (MODE == 1) {
        float fg[16];
        load16f(Aadd + arow, k0, fg);
#pragma unroll
        for (int j = 0; j < 16; ++j) a[j] = (__bf16)(c1 * fa[j] + fg[j]);
      } else if (MODE == 4) {
        float fs[16], fh[16];
        load16f(scale, k0, fs);
        load16f(shift, k0, fh);
#pragma unroll
        for (int j = 0; j < 16; ++j) a[j] = (__bf16)fast_tanh(fa[j] * fs[j] + fh[j]);
      } else {
#pragma unroll
        for (int j = 0; j < 16; ++j) a[j] = (__bf16)fa[j];
      }
    } else {                                 // bf16 source: two 16-byte loads
      const v8bf lo = *(const v8bf*)(A16 + arow + k0);
      const v8bf hi = *(const v8bf*)(A16 + arow + k0 + 16);
#pragma unroll
      for (int j = 0; j < 8; ++j) { a[j] = lo[j]; a[j + 8] = hi[j]; }
    }
    const v16bf b = wlds[(wave * 4 + kt) * 32 + lane];  // two ds_load_b128
    acc = wmma_bf16(a, b, acc);
  }

  float s = 0.f, s2 = 0.f;
  float bcol = 0.0f;
  if (MODE != 0) bcol = bias[col];
#pragma unroll
  for (int r = 0; r < 8; ++r) {
    const unsigned orow = blockIdx.x * 16 + r + 8 * khi; // C: M = r + 8*(l>=16)
    const unsigned o = orow * D + col;
    float v = acc[r];
    if (MODE != 0) v += bcol;               // compile-time guard: no +0.0 adds
    if (MODE == 1)      out16[o] = (__bf16)fast_tanh(v);
    else if (MODE == 2) out16[o] = (__bf16)v;
    else if (MODE == 3) { out32[o] = v; s += v; s2 += v * v; }
    else                out32[o] = v;
  }
  if (MODE == 3) {
    s  += __shfl_xor(s, 16);                // combine the two row-halves
    s2 += __shfl_xor(s2, 16);
    if (khi == 0) {                         // lanes 0..15: one column each
      atomicAdd(&colsum[col], s);
      atomicAdd(&colsumsq[col], s2);
    }
  }
}

// ---- edge pipeline: 128 edges / block, 16 edges / wave ---------------------
// m = tanh(h[src] + normalize(edge_attr)@W_edge + b_edge)   [WMMA K=32]
// PHASE 0: logit = m . att_vec  -> logits[] + atomicMax per dst
// PHASE 1: agg[dst] += m * attn  (fp32 global atomics into L2-resident agg)

template <int PHASE>
__global__ __launch_bounds__(256)
void edge_kernel(const float* __restrict__ edge_attr,
                 const int* __restrict__ src, const int* __restrict__ dst,
                 const float* __restrict__ h,
                 const float* __restrict__ W_edge, const float* __restrict__ b_edge,
                 const float* __restrict__ att_vec,
                 float* __restrict__ logits, unsigned* __restrict__ maxenc,
                 const float* __restrict__ attn, float* __restrict__ agg) {
  __shared__ v16bf wlds[8 * 32];            // 8 KB, fragment-packed (KT=1)
  __shared__ float blds[D];
  __shared__ float alds[D];
  pack_weights<1>(wlds, W_edge);
  for (int i = threadIdx.x; i < D; i += blockDim.x) { blds[i] = b_edge[i]; alds[i] = att_vec[i]; }
  __syncthreads();

  const int lane = threadIdx.x & 31;
  const int wave = threadIdx.x >> 5;
  const int khi  = lane >> 4, ln = lane & 15;
  const int eBase = blockIdx.x * 128 + wave * 16;

  // Normalize edge_attr row and pack the bf16 A fragment (16 edges x K=32).
  const float* earow = edge_attr + (size_t)(eBase + ln) * DE;
  float vals[16];
  load16f(earow, 8 * khi, vals);
  float ss = 0.f;
#pragma unroll
  for (int j = 0; j < 16; ++j) ss += vals[j] * vals[j];
  ss += __shfl_xor(ss, 16);                 // full-row sum of squares
  const float rn = 1.0f / (sqrtf(ss) + 1e-8f);
  v16bf afrag;
#pragma unroll
  for (int j = 0; j < 16; ++j) afrag[j] = (__bf16)(vals[j] * rn);

  // Hoisted per-edge scalars (no shuffles: 16 lanes of a half read the same
  // address -> broadcast-coalesced; 32-bit row offsets -> saddr addressing).
  unsigned hb[8];                           // h row offset per accumulator row
  unsigned ab[8];                           // agg row offset (PHASE 1)
  float    a_r[8];                          // attention weight (PHASE 1)
#pragma unroll
  for (int r = 0; r < 8; ++r) {
    const int e = eBase + r + 8 * khi;
    hb[r] = (unsigned)src[e] * D;
    if (PHASE == 1) { ab[r] = (unsigned)dst[e] * D; a_r[r] = attn[e]; }
  }

  float lacc[8];
#pragma unroll
  for (int r = 0; r < 8; ++r) lacc[r] = 0.f;

  for (int ct = 0; ct < 8; ++ct) {          // 8 column tiles cover D=128
    const v16bf bfrag = wlds[ct * 32 + lane];
    v8f acc = {0.f, 0.f, 0.f, 0.f, 0.f, 0.f, 0.f, 0.f};
    acc = wmma_bf16(afrag, bfrag, acc);

    const int   dcol = ct * 16 + ln;
    const float bcol = blds[dcol];
    const float acol = alds[dcol];

    // Batch the 8 gather loads first so they overlap in flight.
    float hv[8];
#pragma unroll
    for (int r = 0; r < 8; ++r) hv[r] = h[hb[r] + dcol];

    float mv[8];
#pragma unroll
    for (int r = 0; r < 8; ++r) {
      const float m = fast_tanh(acc[r] + bcol + hv[r]);
      if (PHASE == 0) lacc[r] += m * acol;  // partial logit
      else            mv[r] = m * a_r[r];
    }
    if (PHASE == 1) {
#pragma unroll
      for (int r = 0; r < 8; ++r) atomicAdd(&agg[ab[r] + dcol], mv[r]);
    }
  }

  if (PHASE == 0) {
#pragma unroll
    for (int r = 0; r < 8; ++r) {           // reduce 16 lanes of each half
      lacc[r] += __shfl_xor(lacc[r], 1);
      lacc[r] += __shfl_xor(lacc[r], 2);
      lacc[r] += __shfl_xor(lacc[r], 4);
      lacc[r] += __shfl_xor(lacc[r], 8);
    }
    if (ln == 0) {                          // lanes 0 and 16 each own 8 edges
#pragma unroll
      for (int r = 0; r < 8; ++r) {
        const int e = eBase + r + 8 * khi;
        logits[e] = lacc[r];
        atomicMax(&maxenc[dst[e]], enc_f32(lacc[r]));
      }
    }
  }
}

// ---- scalar softmax passes over edges --------------------------------------

__global__ void softmax_exp_kernel(const float* __restrict__ logits,
                                   const int* __restrict__ dst,
                                   const unsigned* __restrict__ maxenc,
                                   float* __restrict__ expv,
                                   float* __restrict__ denom, int E) {
  int e = blockIdx.x * blockDim.x + threadIdx.x;
  if (e >= E) return;
  const int d = dst[e];
  const float ev = __expf(logits[e] - dec_f32(maxenc[d]));
  expv[e] = ev;
  atomicAdd(&denom[d], ev);
}

__global__ void softmax_norm_kernel(const float* __restrict__ expv,
                                    const int* __restrict__ dst,
                                    const float* __restrict__ denom,
                                    float* __restrict__ attn, int E) {
  int e = blockIdx.x * blockDim.x + threadIdx.x;
  if (e >= E) return;
  attn[e] = expv[e] / (denom[dst[e]] + 1e-16f);
}

// ---- batch-norm finalize ----------------------------------------------------

__global__ void bn_finalize_kernel(const float* __restrict__ cs, const float* __restrict__ cs2,
                                   const float* __restrict__ g,  const float* __restrict__ b,
                                   float* __restrict__ scale, float* __restrict__ shift, int N) {
  int i = threadIdx.x;
  if (i < D) {
    const float mu   = cs[i] / (float)N;
    const float var  = cs2[i] / (float)N - mu * mu;   // population variance (jnp.var)
    const float rstd = rsqrtf(var + 1e-5f);
    const float sc   = g[i] * rstd;
    scale[i] = sc;
    shift[i] = b[i] - mu * sc;
  }
}

// ---------------------------------------------------------------------------

extern "C" void kernel_launch(void* const* d_in, const int* in_sizes, int n_in,
                              void* d_out, int out_size, void* d_ws, size_t ws_size,
                              hipStream_t stream) {
  const float* x         = (const float*)d_in[0];
  const float* edge_attr = (const float*)d_in[1];
  const int*   eidx      = (const int*)  d_in[2];
  const float* W_init    = (const float*)d_in[3];
  const float* W_edge    = (const float*)d_in[4];
  const float* b_edge    = (const float*)d_in[5];
  const float* att_vec   = (const float*)d_in[6];
  const float* eps       = (const float*)d_in[7];
  const float* W_u1 = (const float*)d_in[8];  const float* b_u1 = (const float*)d_in[9];
  const float* W_u2 = (const float*)d_in[10]; const float* b_u2 = (const float*)d_in[11];
  const float* W_o1 = (const float*)d_in[12]; const float* b_o1 = (const float*)d_in[13];
  const float* bn_g = (const float*)d_in[14]; const float* bn_b = (const float*)d_in[15];
  const float* W_o2 = (const float*)d_in[16]; const float* b_o2 = (const float*)d_in[17];

  const int N = in_sizes[0] / D;    // 50000 (divisible by 16)
  const int E = in_sizes[1] / DE;   // 800000 (divisible by 128)
  const int* src = eidx;
  const int* dst = eidx + E;

  float* out      = (float*)d_out;            // [N*D] node output
  float* attn_out = out + (size_t)N * D;      // [E]   attention weights

  // ---- workspace carve (≈110 MB) ----
  size_t off = 0;
  auto carve = [&](size_t bytes) -> void* {
    void* p = (char*)d_ws + off;
    off += (bytes + 255) & ~(size_t)255;
    return p;
  };
  float*    h        = (float*)   carve((size_t)N * D * sizeof(float));
  float*    agg      = (float*)   carve((size_t)N * D * sizeof(float));
  float*    z        = (float*)   carve((size_t)N * D * sizeof(float));
  __bf16*   t16      = (__bf16*)  carve((size_t)N * D * sizeof(__bf16));
  __bf16*   u16      = (__bf16*)  carve((size_t)N * D * sizeof(__bf16));
  float*    logits   = (float*)   carve((size_t)E * sizeof(float));
  float*    expv     = (float*)   carve((size_t)E * sizeof(float));
  float*    denom    = (float*)   carve((size_t)N * sizeof(float));
  unsigned* maxenc   = (unsigned*)carve((size_t)N * sizeof(unsigned));
  float*    colsum   = (float*)   carve(D * sizeof(float));
  float*    colsumsq = (float*)   carve(D * sizeof(float));
  float*    scalev   = (float*)   carve(D * sizeof(float));
  float*    shiftv   = (float*)   carve(D * sizeof(float));
  (void)ws_size; (void)n_in; (void)out_size;

  const int mTiles  = N / 16;    // 3125 node M-tiles
  const int eBlocks = E / 128;   // 6250 edge blocks

  // 1) zero/seed accumulators (graph-capture safe; no hipMemset)
  init_kernel<<<(N * D + 255) / 256, 256, 0, stream>>>(agg, denom, maxenc, colsum, colsumsq, N);

  // 2) h = x @ W_init
  gemm128_kernel<0><<<mTiles, 256, 0, stream>>>(x, nullptr, nullptr, W_init, nullptr,
                                                nullptr, nullptr, nullptr,
                                                h, nullptr, nullptr, nullptr);

  // 3) edge pass 1: logits + per-dst max
  edge_kernel<0><<<eBlocks, 256, 0, stream>>>(edge_attr, src, dst, h, W_edge, b_edge, att_vec,
                                              logits, maxenc, attn_out, agg);

  // 4/5) scatter softmax
  softmax_exp_kernel <<<(E + 255) / 256, 256, 0, stream>>>(logits, dst, maxenc, expv, denom, E);
  softmax_norm_kernel<<<(E + 255) / 256, 256, 0, stream>>>(expv, dst, denom, attn_out, E);

  // 6) edge pass 2: recompute m, weighted scatter-add into agg
  edge_kernel<1><<<eBlocks, 256, 0, stream>>>(edge_attr, src, dst, h, W_edge, b_edge, att_vec,
                                              logits, maxenc, attn_out, agg);

  // 7) update MLP: t = tanh(((1+eps)h+agg)@W_u1+b_u1); u = t@W_u2+b_u2
  gemm128_kernel<1><<<mTiles, 256, 0, stream>>>(h, nullptr, agg, W_u1, b_u1,
                                                nullptr, nullptr, eps,
                                                nullptr, t16, nullptr, nullptr);
  gemm128_kernel<2><<<mTiles, 256, 0, stream>>>(nullptr, t16, nullptr, W_u2, b_u2,
                                                nullptr, nullptr, nullptr,
                                                nullptr, u16, nullptr, nullptr);

  // 8) z = u@W_o1+b_o1 with fused batch-norm column statistics
  gemm128_kernel<3><<<mTiles, 256, 0, stream>>>(nullptr, u16, nullptr, W_o1, b_o1,
                                                nullptr, nullptr, nullptr,
                                                z, nullptr, colsum, colsumsq);
  bn_finalize_kernel<<<1, 128, 0, stream>>>(colsum, colsumsq, bn_g, bn_b, scalev, shiftv, N);

  // 9) out = tanh(bn(z))@W_o2 + b_o2
  gemm128_kernel<4><<<mTiles, 256, 0, stream>>>(z, nullptr, nullptr, W_o2, b_o2,
                                                scalev, shiftv, nullptr,
                                                out, nullptr, nullptr, nullptr);
}
